// FastSpeech_30597347017076
// MI455X (gfx1250) — compile-verified, matrix operand order checked by reading, not run on hardware
//
#include <hip/hip_runtime.h>
#include <hip/hip_bf16.h>

// ---------------------------------------------------------------------------
// FastSpeech forward for MI455X (gfx1250, wave32, WMMA).
// GEMMs (QKV, proj, conv1d-as-im2col, mel head) use v_wmma_f32_16x16x32_f16,
// 128x64 block tile, 8 waves x (2x2) subtiles = 4 WMMAs/K-step/wave, double-
// buffered LDS with software pipelining. Weights pre-converted once to f16
// [N][K]; B tiles staged with global_load_async_to_lds_b128 (ASYNCcnt).
// ---------------------------------------------------------------------------

typedef __attribute__((ext_vector_type(16))) _Float16 v16h;
typedef __attribute__((ext_vector_type(8)))  _Float16 v8h;
typedef __attribute__((ext_vector_type(4)))  _Float16 v4h;
typedef __attribute__((ext_vector_type(8)))  float    v8f;

constexpr int Bc   = 16;
constexpr int Lc   = 512;
constexpr int Dc   = 384;
constexpr int Hc   = 2;
constexpr int DHh  = Dc / Hc;      // 192
constexpr int HIDc = 1536;
constexpr int DPc  = 256;
constexpr int NEc  = 4;
constexpr int NDc  = 4;
constexpr int MELc = 80;
constexpr int Tc   = 2560;
constexpr float LN_EPS = 1e-5f;

__device__ __forceinline__ float wred(float v) {
  #pragma unroll
  for (int o = 16; o > 0; o >>= 1) v += __shfl_xor(v, o, 32);
  return v;
}

__device__ __forceinline__ float pe_val(int pos, int ch) {
  const float k = 9.210340371976184f / (float)Dc;   // ln(10000)/D
  float freq = __expf(-(float)(ch & ~1) * k);
  float a = (float)pos * freq;
  return (ch & 1) ? __cosf(a) : __sinf(a);
}

// ISA 7.12.2 16-bit A/B fragment: elems 0..7 -> K = half*8+e ; 8..15 -> 16+half*8+e
__device__ __forceinline__ v16h load_frag(const _Float16* row, int half) {
  const v8h lo = *(const v8h*)(row + half * 8);
  const v8h hi = *(const v8h*)(row + 16 + half * 8);
  return __builtin_shufflevector(lo, hi, 0, 1, 2, 3, 4, 5, 6, 7,
                                 8, 9, 10, 11, 12, 13, 14, 15);
}

// Async global->LDS copy of 16 bytes per lane (GLOBAL_LOAD_ASYNC_TO_LDS_B128,
// tracked by ASYNCcnt). ldsoff is the per-lane LDS byte address.
__device__ __forceinline__ void async_b128(unsigned ldsoff, const _Float16* g) {
  asm volatile("global_load_async_to_lds_b128 %0, %1, off"
               :: "v"(ldsoff), "v"(g) : "memory");
}
__device__ __forceinline__ void wait_async0() {
  asm volatile("s_wait_asynccnt 0" ::: "memory");
}

// ---------------------------------------------------------------------------
// WMMA GEMM: C[M,N] = act( A_gather[M,K] * Wh[N][K]^T + bias )
//   MODE 0: A[m,k] = A[m*K + k]
//   MODE 1: im2col K=3 SAME conv (Cin % 32 == 0, Lrow % 128 == 0: K-tiles
//           never straddle a tap, M-tiles never straddle a batch).
//   GUARDN: bounds-check N (only the MEL head has N % 64 != 0).
// Wh is f16, row-major [N][Kdim] (pre-converted/transposed on device).
// Block 256 = 8 waves; tile 128x64; K step 32; double-buffered LDS; B tiles
// staged via async-to-LDS, A tiles via regs (f32->f16 conversion needed).
// ---------------------------------------------------------------------------
template <int MODE, int RELU, int GUARDN>
__global__ __launch_bounds__(256)
void k_gemm_wmma(const float* __restrict__ A, const _Float16* __restrict__ Wh,
                 const float* __restrict__ bias, float* __restrict__ C,
                 int M, int N, int Kdim, int Cin, int Lrow)
{
  __shared__ _Float16 As[2][128][40];
  __shared__ _Float16 Bt[2][64][40];   // B transposed: [n][k]

  const int tid  = threadIdx.x;
  const int lane = tid & 31;
  const int wv   = tid >> 5;
  const int wmB  = (wv & 3) * 32;
  const int wnB  = (wv >> 2) * 32;
  const int m0   = blockIdx.x * 128;
  const int n0   = blockIdx.y * 64;

  int bb = 0, t0 = 0;
  if (MODE == 1) { bb = m0 / Lrow; t0 = m0 - bb * Lrow; }

  // B staging coords: one v8h (16B) per thread: 256*8 = 2048 = 64x32
  const int bnn = (tid * 8) >> 5, bkk = (tid * 8) & 31;
  const int bgn = n0 + bnn;

  int tap = 0, cinb = 0, kc = 0;       // A im2col tracker (advanced per fetch)
  v4h ah[4];

  auto fetchA = [&](int k0) {          // global -> regs (f32->f16)
    #pragma unroll
    for (int j = 0; j < 4; ++j) {
      int li = (j * 256 + tid) * 4;
      int ml = li >> 5, kk = li & 31;
      float4 v = make_float4(0.f, 0.f, 0.f, 0.f);
      if (MODE == 0) {
        v = *(const float4*)(A + (size_t)(m0 + ml) * Kdim + (k0 + kk));
      } else {
        int t = t0 + ml + tap - 1;
        if (t >= 0 && t < Lrow)
          v = *(const float4*)(A + ((size_t)(bb * Lrow + t)) * Cin + cinb + kk);
      }
      ah[j] = (v4h){(_Float16)v.x, (_Float16)v.y, (_Float16)v.z, (_Float16)v.w};
    }
    if (MODE == 1) { kc += 32; cinb += 32; if (kc == Cin) { kc = 0; cinb = 0; ++tap; } }
  };
  auto storeA = [&](int buf) {         // regs -> LDS
    #pragma unroll
    for (int j = 0; j < 4; ++j) {
      int li = (j * 256 + tid) * 4;
      *(v4h*)&As[buf][li >> 5][li & 31] = ah[j];
    }
  };
  auto issueB = [&](int k0, int buf) { // global f16 -> LDS, async, b128/lane
    if (!GUARDN || bgn < N)
      async_b128((unsigned)(uintptr_t)&Bt[buf][bnn][bkk],
                 Wh + (size_t)bgn * Kdim + (k0 + bkk));
  };

  v8f c[2][2] = {};
  const int r = lane & 15, half = lane >> 4;
  auto compute = [&](int buf) {
    v16h af[2], bf[2];
    #pragma unroll
    for (int mi = 0; mi < 2; ++mi)
      af[mi] = load_frag(&As[buf][wmB + mi * 16 + r][0], half);
    #pragma unroll
    for (int ni = 0; ni < 2; ++ni)
      bf[ni] = load_frag(&Bt[buf][wnB + ni * 16 + r][0], half);
    #pragma unroll
    for (int mi = 0; mi < 2; ++mi)
      #pragma unroll
      for (int ni = 0; ni < 2; ++ni)
        c[mi][ni] = __builtin_amdgcn_wmma_f32_16x16x32_f16(
            false, af[mi], false, bf[ni], (short)0, c[mi][ni], false, false);
  };

  // ---- pipelined main loop: one barrier per K-step ----
  int buf = 0;
  fetchA(0);
  issueB(0, 0);
  storeA(0);
  wait_async0();
  __syncthreads();
  for (int k0 = 0; k0 < Kdim; k0 += 32) {
    bool more = (k0 + 32) < Kdim;
    if (more) { fetchA(k0 + 32); issueB(k0 + 32, buf ^ 1); }
    compute(buf);
    if (more) storeA(buf ^ 1);
    wait_async0();
    __syncthreads();
    buf ^= 1;
  }

  // ---- store (C layout: col = lane&15, row = (lane>>4)*8 + i) ----
  const int cl = lane & 15, rh = (lane >> 4) * 8;
  #pragma unroll
  for (int ni = 0; ni < 2; ++ni) {
    int col = n0 + wnB + ni * 16 + cl;
    if (GUARDN && col >= N) continue;
    float bv = bias ? bias[col] : 0.f;
    #pragma unroll
    for (int mi = 0; mi < 2; ++mi) {
      int rb = m0 + wmB + mi * 16 + rh;
      #pragma unroll
      for (int i = 0; i < 8; ++i) {
        float o = c[mi][ni][i] + bv;
        if (RELU) o = fmaxf(o, 0.f);
        C[(size_t)(rb + i) * N + col] = o;
      }
    }
  }
}

// ---------------------------------------------------------------------------
// Weight conversion: dst (f16, [R][Cd]) from f32 src; trans=1 -> src is [Cd][R].
// ---------------------------------------------------------------------------
__global__ __launch_bounds__(256)
void k_cvt(const float* __restrict__ src, _Float16* __restrict__ dst,
           int R, int Cd, int trans)
{
  int n = blockIdx.y;
  int k = blockIdx.x * 256 + threadIdx.x;
  if (k >= Cd) return;
  float v = trans ? src[(size_t)k * R + n] : src[(size_t)n * Cd + k];
  dst[(size_t)n * Cd + k] = (_Float16)v;
}

// ---------------------------------------------------------------------------
// Embedding + positional encoding.
// ---------------------------------------------------------------------------
__global__ __launch_bounds__(256)
void k_embed(const int* __restrict__ tokens, const float* __restrict__ emb,
             float* __restrict__ x)
{
  int wid  = blockIdx.x * (blockDim.x >> 5) + (threadIdx.x >> 5);
  int lane = threadIdx.x & 31;
  if (wid >= Bc * Lc) return;
  int l = wid % Lc;
  int tok = tokens[wid];
  const float* er = emb + (size_t)tok * Dc;
  float* xr = x + (size_t)wid * Dc;
  #pragma unroll
  for (int e = 0; e < Dc / 32; ++e) {
    int ch = lane + 32 * e;
    xr[ch] = er[ch] + pe_val(l, ch);
  }
}

// ---------------------------------------------------------------------------
// Attention, online softmax, one wave per (b,h,q) row, 8 keys per iteration
// (independent reduction chains; accumulator rescaled once per batch of 8).
// ---------------------------------------------------------------------------
__global__ __launch_bounds__(256)
void k_attn(const float* __restrict__ qkv, float* __restrict__ out, int Lr)
{
  int wid  = blockIdx.x * (blockDim.x >> 5) + (threadIdx.x >> 5);
  int lane = threadIdx.x & 31;
  if (wid >= Bc * Hc * Lr) return;
  int q  = wid % Lr;
  int bh = wid / Lr;
  int h  = bh % Hc, b = bh / Hc;
  const float scale = 0.07216878364870323f;  // 1/sqrt(192)

  const float* base = qkv + (size_t)b * Lr * (3 * Dc);
  const float* qp = base + (size_t)q * (3 * Dc) + h * DHh;
  float qr[6];
  #pragma unroll
  for (int e = 0; e < 6; ++e) qr[e] = qp[lane + 32 * e];

  float m = -1e30f, l = 0.f, acc[6] = {0, 0, 0, 0, 0, 0};
  for (int k0 = 0; k0 < Lr; k0 += 8) {
    float s[8];
    #pragma unroll
    for (int kk = 0; kk < 8; ++kk) {
      const float* kp = base + (size_t)(k0 + kk) * (3 * Dc) + Dc + h * DHh;
      float t = 0.f;
      #pragma unroll
      for (int e = 0; e < 6; ++e) t += qr[e] * kp[lane + 32 * e];
      s[kk] = t;
    }
    #pragma unroll
    for (int kk = 0; kk < 8; ++kk) s[kk] = wred(s[kk]) * scale;
    float mx = s[0];
    #pragma unroll
    for (int kk = 1; kk < 8; ++kk) mx = fmaxf(mx, s[kk]);
    float mn = fmaxf(m, mx);
    float corr = __expf(m - mn);
    l *= corr;
    #pragma unroll
    for (int e = 0; e < 6; ++e) acc[e] *= corr;
    #pragma unroll
    for (int kk = 0; kk < 8; ++kk) {
      float pc = __expf(s[kk] - mn);
      l += pc;
      const float* vp = base + (size_t)(k0 + kk) * (3 * Dc) + 2 * Dc + h * DHh;
      #pragma unroll
      for (int e = 0; e < 6; ++e) acc[e] = fmaf(pc, vp[lane + 32 * e], acc[e]);
    }
    m = mn;
  }
  float inv = 1.f / l;
  float* op = out + ((size_t)b * Lr + q) * Dc + h * DHh;
  #pragma unroll
  for (int e = 0; e < 6; ++e) op[lane + 32 * e] = acc[e] * inv;
}

// ---------------------------------------------------------------------------
// LayerNorm (N % 32 == 0, N <= 384); optional residual r; in-place safe.
// ---------------------------------------------------------------------------
__global__ __launch_bounds__(256)
void k_layernorm(const float* __restrict__ a, const float* __restrict__ r,
                 const float* __restrict__ g, const float* __restrict__ be,
                 float* __restrict__ o, int N, int rows)
{
  int wid  = blockIdx.x * (blockDim.x >> 5) + (threadIdx.x >> 5);
  int lane = threadIdx.x & 31;
  if (wid >= rows) return;
  int npl = N >> 5;
  float xv[12];
  const float* ar = a + (size_t)wid * N;
  const float* rr = r ? r + (size_t)wid * N : nullptr;
  float s = 0.f;
  for (int e = 0; e < npl; ++e) {
    int ch = lane + 32 * e;
    float v = ar[ch] + (rr ? rr[ch] : 0.f);
    xv[e] = v;
    s += v;
  }
  float mean = wred(s) / (float)N;
  float s2 = 0.f;
  for (int e = 0; e < npl; ++e) { float d = xv[e] - mean; s2 += d * d; }
  float rstd = __frsqrt_rn(wred(s2) / (float)N + LN_EPS);
  float* orow = o + (size_t)wid * N;
  for (int e = 0; e < npl; ++e) {
    int ch = lane + 32 * e;
    orow[ch] = (xv[e] - mean) * rstd * g[ch] + be[ch];
  }
}

__global__ __launch_bounds__(256)
void k_durpred(const float* __restrict__ h, const float* __restrict__ lw,
               const float* __restrict__ lb, float* __restrict__ out)
{
  int wid  = blockIdx.x * (blockDim.x >> 5) + (threadIdx.x >> 5);
  int lane = threadIdx.x & 31;
  if (wid >= Bc * Lc) return;
  float s = 0.f;
  #pragma unroll
  for (int e = 0; e < DPc / 32; ++e) {
    int ch = lane + 32 * e;
    s += h[(size_t)wid * DPc + ch] * lw[ch];
  }
  s = wred(s);
  if (lane == 0) out[wid] = s + lb[0];
}

__global__ __launch_bounds__(512)
void k_scan(const int* __restrict__ dur, const int* __restrict__ len,
            int* __restrict__ cum)
{
  __shared__ int s[Lc];
  int b = blockIdx.x, t = threadIdx.x;
  int lb = len[b];
  s[t] = (t < lb) ? dur[b * Lc + t] : 0;
  __syncthreads();
  for (int off = 1; off < Lc; off <<= 1) {
    int v = (t >= off) ? s[t - off] : 0;
    __syncthreads();
    s[t] += v;
    __syncthreads();
  }
  cum[b * Lc + t] = s[t];
}

__global__ __launch_bounds__(256)
void k_lenreg(const float* __restrict__ x, const int* __restrict__ cum,
              float* __restrict__ y)
{
  int wid  = blockIdx.x * (blockDim.x >> 5) + (threadIdx.x >> 5);
  int lane = threadIdx.x & 31;
  if (wid >= Bc * Tc) return;
  int b = wid / Tc, t = wid % Tc;
  const int* c = cum + b * Lc;
  int total = c[Lc - 1];
  int lo = 0, hi = Lc;
  while (lo < hi) { int mid = (lo + hi) >> 1; if (c[mid] <= t) lo = mid + 1; else hi = mid; }
  int idx = lo < Lc ? lo : Lc - 1;
  bool valid = t < total;
  const float* xr = x + ((size_t)b * Lc + idx) * Dc;
  float* yr = y + ((size_t)b * Tc + t) * Dc;
  #pragma unroll
  for (int e = 0; e < Dc / 32; ++e) {
    int ch = lane + 32 * e;
    float v = valid ? xr[ch] : 0.f;
    yr[ch] = v + pe_val(t, ch);
  }
}

// ---------------------------------------------------------------------------
// Host orchestration
// ---------------------------------------------------------------------------
extern "C" void kernel_launch(void* const* d_in, const int* in_sizes, int n_in,
                              void* d_out, int out_size, void* d_ws, size_t ws_size,
                              hipStream_t stream) {
  (void)in_sizes; (void)n_in; (void)out_size; (void)ws_size;

  const int*   tokens   = (const int*)d_in[0];
  const int*   lengths  = (const int*)d_in[1];
  const int*   durs     = (const int*)d_in[2];
  const float* emb      = (const float*)d_in[3];
  enum { WQKV = 0, BQKV, WO, BO, LN1G, LN1B, C1W, C1B, C2W, C2B, LN2G, LN2B };
  const float* enc[12]; const float* dec[12];
  for (int i = 0; i < 12; ++i) { enc[i] = (const float*)d_in[4 + i];
                                 dec[i] = (const float*)d_in[16 + i]; }
  const float* out_w = (const float*)d_in[28];
  const float* out_b = (const float*)d_in[29];
  const float* dpc1w = (const float*)d_in[30];
  const float* dpc1b = (const float*)d_in[31];
  const float* dpl1g = (const float*)d_in[32];
  const float* dpl1b = (const float*)d_in[33];
  const float* dpc2w = (const float*)d_in[34];
  const float* dpc2b = (const float*)d_in[35];
  const float* dpl2g = (const float*)d_in[36];
  const float* dpl2b = (const float*)d_in[37];
  const float* dplw  = (const float*)d_in[38];
  const float* dplb  = (const float*)d_in[39];

  const size_t strd[12] = {
    (size_t)3 * Dc * Dc, (size_t)3 * Dc, (size_t)Dc * Dc, (size_t)Dc,
    (size_t)Dc, (size_t)Dc,
    (size_t)3 * Dc * HIDc, (size_t)HIDc, (size_t)3 * HIDc * Dc, (size_t)Dc,
    (size_t)Dc, (size_t)Dc };

  // ---- workspace carve ----
  float* p = (float*)d_ws;
  float* bufX   = p; p += (size_t)Bc * Tc * Dc;
  float* bufA   = p; p += (size_t)Bc * Tc * Dc;
  float* bufB   = p; p += (size_t)Bc * Tc * Dc;
  float* bufBig = p; p += (size_t)Bc * Tc * HIDc;
  float* bufD1  = p; p += (size_t)Bc * Lc * DPc;
  float* bufD2  = p; p += (size_t)Bc * Lc * DPc;
  int*   cum    = (int*)p; p += Bc * Lc;
  // f16 weight shadow region, [N][K] row-major per matrix
  _Float16* hp = (_Float16*)p;
  _Float16 *wqkvh[8], *woh[8], *c1wh[8], *c2wh[8];
  for (int i = 0; i < 8; ++i) {
    wqkvh[i] = hp; hp += (size_t)3 * Dc * Dc;        // [1152][384]
    woh[i]   = hp; hp += (size_t)Dc * Dc;            // [384][384]
    c1wh[i]  = hp; hp += (size_t)HIDc * 3 * Dc;      // [1536][1152]
    c2wh[i]  = hp; hp += (size_t)Dc * 3 * HIDc;      // [384][4608]
  }
  _Float16* dpc1h = hp; hp += (size_t)DPc * 3 * Dc;  // [256][1152]
  _Float16* dpc2h = hp; hp += (size_t)DPc * 3 * DPc; // [256][768]
  _Float16* outwh = hp; hp += (size_t)MELc * Dc;     // [80][384]

  float* mel_out = (float*)d_out;
  float* dur_out = mel_out + (size_t)Bc * Tc * MELc;

  auto cvt = [&](const float* src, _Float16* dst, int R, int Cd, int trans) {
    dim3 g((Cd + 255) / 256, R);
    k_cvt<<<g, 256, 0, stream>>>(src, dst, R, Cd, trans);
  };
  // one-time weight conversion (wqkv/wo already [N][K]; convs/out_w transposed)
  for (int i = 0; i < 8; ++i) {
    const float* const* w = (i < NEc) ? enc : dec;
    int l = (i < NEc) ? i : i - NEc;
    cvt(w[WQKV] + l * strd[WQKV], wqkvh[i], 3 * Dc, Dc, 0);
    cvt(w[WO]   + l * strd[WO],   woh[i],   Dc, Dc, 0);
    cvt(w[C1W]  + l * strd[C1W],  c1wh[i],  HIDc, 3 * Dc, 1);
    cvt(w[C2W]  + l * strd[C2W],  c2wh[i],  Dc, 3 * HIDc, 1);
  }
  cvt(dpc1w, dpc1h, DPc, 3 * Dc, 1);
  cvt(dpc2w, dpc2h, DPc, 3 * DPc, 1);
  cvt(out_w, outwh, MELc, Dc, 1);

  auto grid = [](int M, int N) { return dim3((M + 127) / 128, (N + 63) / 64); };
  auto gemm = [&](const float* A, const _Float16* Wh, const float* bias,
                  float* C, int M, int N, int Kd) {
    k_gemm_wmma<0, 0, 0><<<grid(M, N), 256, 0, stream>>>(A, Wh, bias, C, M, N, Kd, 0, 0);
  };
  auto gemm_mel = [&](const float* A, const _Float16* Wh, const float* bias,
                      float* C, int M, int N, int Kd) {
    k_gemm_wmma<0, 0, 1><<<grid(M, N), 256, 0, stream>>>(A, Wh, bias, C, M, N, Kd, 0, 0);
  };
  auto conv_relu = [&](const float* A, const _Float16* Wh, const float* bias,
                       float* C, int M, int Cout, int Cin, int Lrow) {
    k_gemm_wmma<1, 1, 0><<<grid(M, Cout), 256, 0, stream>>>(A, Wh, bias, C, M, Cout,
                                                            3 * Cin, Cin, Lrow);
  };
  auto conv_lin = [&](const float* A, const _Float16* Wh, const float* bias,
                      float* C, int M, int Cout, int Cin, int Lrow) {
    k_gemm_wmma<1, 0, 0><<<grid(M, Cout), 256, 0, stream>>>(A, Wh, bias, C, M, Cout,
                                                            3 * Cin, Cin, Lrow);
  };
  auto ln = [&](const float* a, const float* r, const float* g,
                const float* be, float* o, int N, int rows) {
    k_layernorm<<<(rows + 7) / 8, 256, 0, stream>>>(a, r, g, be, o, N, rows);
  };
  auto fft_block = [&](const float* const* w, int l, int wi, float* x,
                       float* t1, float* t2, int rows, int Lrow) {
    gemm(x, wqkvh[wi], w[BQKV] + l * strd[BQKV], bufBig, rows, 3 * Dc, Dc);
    int aw = Bc * Hc * Lrow;
    k_attn<<<(aw + 7) / 8, 256, 0, stream>>>(bufBig, t1, Lrow);
    gemm(t1, woh[wi], w[BO] + l * strd[BO], t2, rows, Dc, Dc);
    ln(x, t2, w[LN1G] + l * strd[LN1G], w[LN1B] + l * strd[LN1B], x, Dc, rows);
    conv_relu(x, c1wh[wi], w[C1B] + l * strd[C1B], bufBig, rows, HIDc, Dc, Lrow);
    conv_lin(bufBig, c2wh[wi], w[C2B] + l * strd[C2B], t2, rows, Dc, HIDc, Lrow);
    ln(x, t2, w[LN2G] + l * strd[LN2G], w[LN2B] + l * strd[LN2B], x, Dc, rows);
  };

  const int rowsE = Bc * Lc;
  const int rowsD = Bc * Tc;

  // ---- encoder ----
  k_embed<<<(rowsE + 7) / 8, 256, 0, stream>>>(tokens, emb, bufX);
  for (int i = 0; i < NEc; ++i)
    fft_block(enc, i, i, bufX, bufA, bufB, rowsE, Lc);

  // ---- duration predictor ----
  conv_relu(bufX, dpc1h, dpc1b, bufD1, rowsE, DPc, Dc, Lc);
  ln(bufD1, nullptr, dpl1g, dpl1b, bufD1, DPc, rowsE);
  conv_relu(bufD1, dpc2h, dpc2b, bufD2, rowsE, DPc, DPc, Lc);
  ln(bufD2, nullptr, dpl2g, dpl2b, bufD2, DPc, rowsE);
  k_durpred<<<(rowsE + 7) / 8, 256, 0, stream>>>(bufD2, dplw, dplb, dur_out);

  // ---- length regulate + decoder PE ----
  k_scan<<<Bc, Lc, 0, stream>>>(durs, lengths, cum);
  k_lenreg<<<(rowsD + 7) / 8, 256, 0, stream>>>(bufX, cum, bufA);

  // ---- decoder ----
  for (int i = 0; i < NDc; ++i)
    fft_block(dec, i, NEc + i, bufA, bufB, bufX, rowsD, Tc);

  // ---- mel head ----
  gemm_mel(bufA, outwh, out_b, mel_out, rowsD, MELc, Dc);
}
